// SwinTransformerEncoderBlock_2259152798424
// MI455X (gfx1250) — compile-verified, matrix operand order checked by reading, not run on hardware
//
#include <hip/hip_runtime.h>
#include <hip/hip_bf16.h>

// ---------------------------------------------------------------------------
// Types for CDNA5 WMMA (wave32, 16x16x32 f16 -> f32)
// ---------------------------------------------------------------------------
typedef _Float16 h16;
typedef __attribute__((ext_vector_type(16))) _Float16 v16h;
typedef __attribute__((ext_vector_type(8)))  _Float16 v8h;
typedef __attribute__((ext_vector_type(4)))  _Float16 v4h;
typedef __attribute__((ext_vector_type(8)))  float    v8f;

static __device__ __forceinline__ v16h cat8(v8h lo, v8h hi) {
  return __builtin_shufflevector(lo, hi, 0,1,2,3,4,5,6,7,8,9,10,11,12,13,14,15);
}
static __device__ __forceinline__ v8f wmma_f16(v16h a, v16h b, v8f c) {
  // D = A(16x32) * B(32x16) + C(16x16), f32 accumulate
  return __builtin_amdgcn_wmma_f32_16x16x32_f16(false, a, false, b, (short)0, c, false, false);
}

// Swin static config (matches reference)
#define SW_H   64
#define SW_W   64
#define SW_WIN 8
#define SW_C   512
#define SW_L   64
#define SW_B   8
#define SW_M   32768   // B * H * W  (rows in window layout)

// window-row -> image-row with cyclic shift s (used for reverse shift/unpartition)
static __device__ __forceinline__ int unshift_row(int m, int s) {
  int b = m >> 12;
  int w = (m >> 6) & 63, t = m & 63;
  int qy = ((w >> 3) << 3) + (t >> 3);
  int qx = ((w & 7) << 3) + (t & 7);
  int py = (qy + s) & 63, px = (qx + s) & 63;
  return (b << 12) + py * 64 + px;
}

// ---------------------------------------------------------------------------
// Generic WMMA GEMM:  out[M,N] = A[M,K](f16) * B[K,N](f16)  (+ epilogue)
// Block tile 128x64, 8 waves (4x2), each wave 32x32 (2x2 WMMA tiles), K-step 32.
// epi: 0 = +bias -> f16
//      1 = +bias +res(f32, stride N) -> f32
//      2 = gelu(+bias) -> f16
//      3 = +bias +res -> f32 stored at permuted row unshift_row(row, shift)
//      4 = plain -> f32
// ---------------------------------------------------------------------------
__global__ __launch_bounds__(256) void gemm_wmma(
    const h16* __restrict__ A, const h16* __restrict__ B,
    const float* __restrict__ bias, const float* __restrict__ res,
    void* __restrict__ out, int M, int N, int K, int epi, int shift)
{
  __shared__ h16 As[128][32];
  __shared__ h16 Bst[64][32];   // B tile transposed: Bst[n][k]

  const int tid   = threadIdx.x;
  const int mBase = blockIdx.x * 128;
  const int nBase = blockIdx.y * 64;
  const int wv    = tid >> 5, lane = tid & 31;
  const int wm    = wv & 3,  wn   = wv >> 2;
  const int l15   = lane & 15, lh = lane >> 4;
  const int khalf = lh * 8;

  v8f acc00 = {}, acc01 = {}, acc10 = {}, acc11 = {};

  const int ar = tid >> 1, ac = (tid & 1) * 16;   // A loader: 2 thr/row, 16 halfs each
  const int bk = tid >> 3, bn = (tid & 7) * 8;    // B loader: 8 halfs, transpose-store

  for (int kt = 0; kt < K; kt += 32) {
    const v8h* ap = (const v8h*)(A + (size_t)(mBase + ar) * K + kt + ac);
    v8h av0 = ap[0], av1 = ap[1];
    v8h bvv = *(const v8h*)(B + (size_t)(kt + bk) * N + nBase + bn);

    __syncthreads();
    *(v8h*)&As[ar][ac]     = av0;
    *(v8h*)&As[ar][ac + 8] = av1;
    #pragma unroll
    for (int j = 0; j < 8; ++j) Bst[bn + j][bk] = bvv[j];
    __syncthreads();

    const int m0 = wm * 32 + l15;
    v16h a0 = cat8(*(const v8h*)&As[m0][khalf],      *(const v8h*)&As[m0][16 + khalf]);
    v16h a1 = cat8(*(const v8h*)&As[m0 + 16][khalf], *(const v8h*)&As[m0 + 16][16 + khalf]);
    const int n0 = wn * 32 + l15;
    v16h b0 = cat8(*(const v8h*)&Bst[n0][khalf],      *(const v8h*)&Bst[n0][16 + khalf]);
    v16h b1 = cat8(*(const v8h*)&Bst[n0 + 16][khalf], *(const v8h*)&Bst[n0 + 16][16 + khalf]);

    acc00 = wmma_f16(a0, b0, acc00);
    acc01 = wmma_f16(a0, b1, acc01);
    acc10 = wmma_f16(a1, b0, acc10);
    acc11 = wmma_f16(a1, b1, acc11);
  }

  v8f accs[2][2] = {{acc00, acc01}, {acc10, acc11}};
  #pragma unroll
  for (int i = 0; i < 2; ++i)
    #pragma unroll
    for (int j = 0; j < 2; ++j)
      #pragma unroll
      for (int r = 0; r < 8; ++r) {
        int row = mBase + wm * 32 + i * 16 + lh * 8 + r;
        int col = nBase + wn * 32 + j * 16 + l15;
        float v = accs[i][j][r];
        if (epi != 4) v += bias[col];
        if (epi == 1 || epi == 3) v += res[(size_t)row * N + col];
        if (epi == 2) v = 0.5f * v * (1.0f + erff(v * 0.70710678118654752f));
        if (epi == 0 || epi == 2) {
          ((h16*)out)[(size_t)row * N + col] = (h16)v;
        } else {
          int drow = (epi == 3) ? unshift_row(row, shift) : row;
          ((float*)out)[(size_t)drow * N + col] = v;
        }
      }
}

// ---------------------------------------------------------------------------
// Fused window attention: one block per (window, head). 4 waves.
// S = (Q K^T)/sqrt(hd) + relposbias (+shift mask); P = softmax(S); O = P V.
// Q/K/V read from qkv f16 [32768, 1536]; O written f16 [32768, 512].
// ---------------------------------------------------------------------------
__global__ __launch_bounds__(128) void attn_wmma(
    const h16* __restrict__ qkv, const float* __restrict__ tab,
    h16* __restrict__ out, int shifted)
{
  __shared__ h16 qs[64][32];
  __shared__ h16 ks[64][32];
  __shared__ h16 vt[32][64];      // V transposed: vt[hd][key]
  __shared__ h16 ps[4][16][64];   // per-wave probability strip

  const int win  = blockIdx.x;    // b*64 + wy*8 + wx
  const int head = blockIdx.y;
  const int tid  = threadIdx.x;

  // cooperative load of Q, K, V (64x32 each)
  {
    int row = tid >> 1;
    int c0  = (tid & 1) * 16;
    size_t base = (size_t)(win * 64 + row) * 1536 + head * 32 + c0;
    v8h q0 = *(const v8h*)(qkv + base),        q1 = *(const v8h*)(qkv + base + 8);
    v8h k0 = *(const v8h*)(qkv + base + 512),  k1 = *(const v8h*)(qkv + base + 520);
    v8h v0 = *(const v8h*)(qkv + base + 1024), v1 = *(const v8h*)(qkv + base + 1032);
    *(v8h*)&qs[row][c0] = q0; *(v8h*)&qs[row][c0 + 8] = q1;
    *(v8h*)&ks[row][c0] = k0; *(v8h*)&ks[row][c0 + 8] = k1;
    #pragma unroll
    for (int j = 0; j < 8; ++j) { vt[c0 + j][row] = v0[j]; vt[c0 + 8 + j][row] = v1[j]; }
  }
  __syncthreads();

  const int wv = tid >> 5, lane = tid & 31;
  const int l15 = lane & 15, lh = lane >> 4;
  const int khalf = lh * 8;

  // S strip: rows 16*wv .. 16*wv+15, all 64 keys
  v16h aq = cat8(*(const v8h*)&qs[wv * 16 + l15][khalf],
                 *(const v8h*)&qs[wv * 16 + l15][16 + khalf]);
  v8f s[4];
  #pragma unroll
  for (int nt = 0; nt < 4; ++nt) {
    v16h bk = cat8(*(const v8h*)&ks[nt * 16 + l15][khalf],
                   *(const v8h*)&ks[nt * 16 + l15][16 + khalf]);
    v8f z = {};
    s[nt] = wmma_f16(aq, bk, z);
  }

  const float sc = 0.17677669529663687f;  // 1/sqrt(32)
  const int wloc = win & 63;
  const int wy = (wloc >> 3) << 3, wx = (wloc & 7) << 3;

  float vals[4][8];
  #pragma unroll
  for (int nt = 0; nt < 4; ++nt)
    #pragma unroll
    for (int r = 0; r < 8; ++r) {
      int qrow = wv * 16 + lh * 8 + r;
      int kcol = nt * 16 + l15;
      int dy = (qrow >> 3) - (kcol >> 3) + 7;
      int dx = (qrow & 7) - (kcol & 7) + 7;
      float val = s[nt][r] * sc + tab[dy * 15 + dx];
      if (shifted) {
        int qy = wy + (qrow >> 3), qx = wx + (qrow & 7);
        int ky = wy + (kcol >> 3), kx = wx + (kcol & 7);
        int rq = (qy < 56 ? 0 : (qy < 60 ? 1 : 2)) * 3 + (qx < 56 ? 0 : (qx < 60 ? 1 : 2));
        int rk = (ky < 56 ? 0 : (ky < 60 ? 1 : 2)) * 3 + (kx < 56 ? 0 : (kx < 60 ? 1 : 2));
        if (rq != rk) val += -1e9f;
      }
      vals[nt][r] = val;
    }

  // row-wise softmax (each row lives across the 16 lanes of a half-wave)
  #pragma unroll
  for (int r = 0; r < 8; ++r) {
    float mx = fmaxf(fmaxf(vals[0][r], vals[1][r]), fmaxf(vals[2][r], vals[3][r]));
    #pragma unroll
    for (int off = 1; off < 16; off <<= 1) mx = fmaxf(mx, __shfl_xor(mx, off, 32));
    float sum = 0.f;
    #pragma unroll
    for (int nt = 0; nt < 4; ++nt) { float e = expf(vals[nt][r] - mx); vals[nt][r] = e; sum += e; }
    #pragma unroll
    for (int off = 1; off < 16; off <<= 1) sum += __shfl_xor(sum, off, 32);
    float inv = 1.f / sum;
    int prow = lh * 8 + r;
    #pragma unroll
    for (int nt = 0; nt < 4; ++nt) ps[wv][prow][nt * 16 + l15] = (h16)(vals[nt][r] * inv);
  }
  __syncthreads();

  // O = P(16x64) * V(64x32): two K steps, two N tiles
  v8f o0 = {}, o1 = {};
  #pragma unroll
  for (int ks2 = 0; ks2 < 2; ++ks2) {
    v16h apf = cat8(*(const v8h*)&ps[wv][l15][ks2 * 32 + khalf],
                    *(const v8h*)&ps[wv][l15][ks2 * 32 + 16 + khalf]);
    v16h bv0 = cat8(*(const v8h*)&vt[l15][ks2 * 32 + khalf],
                    *(const v8h*)&vt[l15][ks2 * 32 + 16 + khalf]);
    v16h bv1 = cat8(*(const v8h*)&vt[16 + l15][ks2 * 32 + khalf],
                    *(const v8h*)&vt[16 + l15][ks2 * 32 + 16 + khalf]);
    o0 = wmma_f16(apf, bv0, o0);
    o1 = wmma_f16(apf, bv1, o1);
  }
  #pragma unroll
  for (int r = 0; r < 8; ++r) {
    size_t orow = (size_t)(win * 64 + wv * 16 + lh * 8 + r) * 512 + head * 32;
    out[orow + l15]      = (h16)o0[r];
    out[orow + 16 + l15] = (h16)o1[r];
  }
}

// ---------------------------------------------------------------------------
// LayerNorm over C=512 with optional window-partition(+shift) gather and
// optional f32 copy-out of the gathered row (pre-LN residual).
// shift >= 0: src row gathered from image layout with cyclic shift `shift`.
// shift <  0: identity rows.
// ---------------------------------------------------------------------------
__global__ __launch_bounds__(128) void ln512_kernel(
    const float* __restrict__ src, const float* __restrict__ g,
    const float* __restrict__ b, h16* __restrict__ outh,
    float* __restrict__ outcopy, int shift)
{
  __shared__ float sb[8];
  const int m = blockIdx.x;
  int srow = m;
  if (shift >= 0) {
    int bi = m >> 12;
    int w = (m >> 6) & 63, t = m & 63;
    int qy = ((w >> 3) << 3) + (t >> 3);
    int qx = ((w & 7) << 3) + (t & 7);
    srow = (bi << 12) + ((qy + shift) & 63) * 64 + ((qx + shift) & 63);
  }
  const int t = threadIdx.x;           // 128 threads * 4 floats
  float4 x = ((const float4*)(src + (size_t)srow * 512))[t];

  float lsum = x.x + x.y + x.z + x.w;
  float lsq  = x.x * x.x + x.y * x.y + x.z * x.z + x.w * x.w;
  #pragma unroll
  for (int off = 16; off; off >>= 1) {
    lsum += __shfl_xor(lsum, off, 32);
    lsq  += __shfl_xor(lsq,  off, 32);
  }
  if ((t & 31) == 0) { sb[(t >> 5) * 2] = lsum; sb[(t >> 5) * 2 + 1] = lsq; }
  __syncthreads();
  float ts = 0.f, tq = 0.f;
  #pragma unroll
  for (int i = 0; i < 4; ++i) { ts += sb[i * 2]; tq += sb[i * 2 + 1]; }

  float mean = ts * (1.f / 512.f);
  float var  = tq * (1.f / 512.f) - mean * mean;
  float rs   = rsqrtf(var + 1e-5f);

  float4 gg = ((const float4*)g)[t];
  float4 bb = ((const float4*)b)[t];
  v4h o;
  o[0] = (h16)((x.x - mean) * rs * gg.x + bb.x);
  o[1] = (h16)((x.y - mean) * rs * gg.y + bb.y);
  o[2] = (h16)((x.z - mean) * rs * gg.z + bb.z);
  o[3] = (h16)((x.w - mean) * rs * gg.w + bb.w);
  *(v4h*)(outh + (size_t)m * 512 + t * 4) = o;
  if (outcopy) ((float4*)(outcopy + (size_t)m * 512))[t] = x;
}

// ---------------------------------------------------------------------------
// PatchMerge gather (2x2 pixels -> 4C) + LayerNorm over 2048, f16 out.
// Feature order per reference transpose: f = w_sub*1024 + h_sub*512 + c.
// ---------------------------------------------------------------------------
__global__ __launch_bounds__(256) void ln_merge_kernel(
    const float* __restrict__ src, const float* __restrict__ g,
    const float* __restrict__ b, h16* __restrict__ outh)
{
  __shared__ float sb[16];
  const int m = blockIdx.x;                 // 0..8191
  const int bi = m >> 10, p = m & 1023;
  const int h2 = p >> 5, w2 = p & 31;
  const int t = threadIdx.x;                // 256 threads * 8 floats
  const int f = t * 8;
  const int wsub = f >> 10, hsub = (f >> 9) & 1, c = f & 511;
  const int srow = (bi << 12) + (h2 * 2 + hsub) * 64 + (w2 * 2 + wsub);
  const float4* sp = (const float4*)(src + (size_t)srow * 512 + c);
  float4 x0 = sp[0], x1 = sp[1];

  float lsum = x0.x + x0.y + x0.z + x0.w + x1.x + x1.y + x1.z + x1.w;
  float lsq  = x0.x*x0.x + x0.y*x0.y + x0.z*x0.z + x0.w*x0.w
             + x1.x*x1.x + x1.y*x1.y + x1.z*x1.z + x1.w*x1.w;
  #pragma unroll
  for (int off = 16; off; off >>= 1) {
    lsum += __shfl_xor(lsum, off, 32);
    lsq  += __shfl_xor(lsq,  off, 32);
  }
  if ((t & 31) == 0) { sb[(t >> 5) * 2] = lsum; sb[(t >> 5) * 2 + 1] = lsq; }
  __syncthreads();
  float ts = 0.f, tq = 0.f;
  #pragma unroll
  for (int i = 0; i < 8; ++i) { ts += sb[i * 2]; tq += sb[i * 2 + 1]; }

  float mean = ts * (1.f / 2048.f);
  float var  = tq * (1.f / 2048.f) - mean * mean;
  float rs   = rsqrtf(var + 1e-5f);

  float4 g0 = ((const float4*)(g + f))[0], g1 = ((const float4*)(g + f))[1];
  float4 b0 = ((const float4*)(b + f))[0], b1 = ((const float4*)(b + f))[1];
  h16* op = outh + (size_t)m * 2048 + f;
  v4h oa, ob;
  oa[0] = (h16)((x0.x - mean) * rs * g0.x + b0.x);
  oa[1] = (h16)((x0.y - mean) * rs * g0.y + b0.y);
  oa[2] = (h16)((x0.z - mean) * rs * g0.z + b0.z);
  oa[3] = (h16)((x0.w - mean) * rs * g0.w + b0.w);
  ob[0] = (h16)((x1.x - mean) * rs * g1.x + b1.x);
  ob[1] = (h16)((x1.y - mean) * rs * g1.y + b1.y);
  ob[2] = (h16)((x1.z - mean) * rs * g1.z + b1.z);
  ob[3] = (h16)((x1.w - mean) * rs * g1.w + b1.w);
  *(v4h*)op = oa;
  *(v4h*)(op + 4) = ob;
}

// f32 -> f16 weight convert, dst is [K][N]; transpose=1 means src is [N][K].
__global__ __launch_bounds__(256) void cvt_kernel(
    const float* __restrict__ src, h16* __restrict__ dst, int K, int N, int trans)
{
  int i = blockIdx.x * 256 + threadIdx.x;
  if (i >= K * N) return;
  int k = i / N, n = i - k * N;
  float v = trans ? src[(size_t)n * K + k] : src[(size_t)k * N + n];
  dst[i] = (h16)v;
}

// ---------------------------------------------------------------------------
// Host-side orchestration
// ---------------------------------------------------------------------------
extern "C" void kernel_launch(void* const* d_in, const int* in_sizes, int n_in,
                              void* d_out, int out_size, void* d_ws, size_t ws_size,
                              hipStream_t stream) {
  (void)in_sizes; (void)n_in; (void)out_size; (void)ws_size;
  const float* x   = (const float*)d_in[0];
  const float* ipw = (const float*)d_in[1];
  const float* ipb = (const float*)d_in[2];
  const float* ow  = (const float*)d_in[3];
  const float* ob  = (const float*)d_in[4];
  const float* tab = (const float*)d_in[5];
  const float* g1  = (const float*)d_in[6];
  const float* b1  = (const float*)d_in[7];
  const float* g2  = (const float*)d_in[8];
  const float* b2  = (const float*)d_in[9];
  const float* w1  = (const float*)d_in[10];
  const float* bb1 = (const float*)d_in[11];
  const float* w2  = (const float*)d_in[12];
  const float* bb2 = (const float*)d_in[13];
  const float* mg  = (const float*)d_in[14];
  const float* mb  = (const float*)d_in[15];
  const float* mw  = (const float*)d_in[16];

  char* ws = (char*)d_ws;
  const size_t MB = 1ull << 20;
  float* xw     = (float*)(ws);               //  64 MB window-layout pre-LN residual
  float* xa     = (float*)(ws + 64  * MB);    //  64 MB post-attention residual
  float* ximg   = (float*)(ws + 128 * MB);    //  64 MB image-layout layer output
  h16*   hbuf   = (h16*)  (ws + 192 * MB);    //  32 MB LN outputs (f16)
  h16*   qkvb   = (h16*)  (ws + 224 * MB);    //  96 MB qkv f16
  h16*   attn_o = (h16*)  (ws + 320 * MB);    //  32 MB attention output f16
  h16*   gbuf   = (h16*)  (ws + 224 * MB);    // 128 MB GELU acts (reuses qkv+attn region)
  h16*   wbuf   = (h16*)  (ws + 352 * MB);    //   8 MB converted weights

  for (int d = 0; d < 2; ++d) {
    const float* xin = (d == 0) ? x : ximg;
    const int shift = (d == 1) ? 4 : 0;

    // LN1 + window partition (+shift); keep gathered f32 copy as residual
    ln512_kernel<<<SW_M, 128, 0, stream>>>(xin, g1 + d * 512, b1 + d * 512,
                                           hbuf, xw, shift);

    // QKV = h @ iw^T + ib   (iw is [1536,512] -> transpose to [K=512][N=1536])
    cvt_kernel<<<(512 * 1536 + 255) / 256, 256, 0, stream>>>(
        ipw + (size_t)d * 1536 * 512, wbuf, 512, 1536, 1);
    gemm_wmma<<<dim3(SW_M / 128, 1536 / 64), 256, 0, stream>>>(
        hbuf, wbuf, ipb + d * 1536, nullptr, qkvb, SW_M, 1536, 512, /*epi*/0, 0);

    // fused window attention
    attn_wmma<<<dim3(512, 16), 128, 0, stream>>>(qkvb, tab + d * 225, attn_o, d == 1);

    // out projection + residual(xw) -> xa (f32)
    cvt_kernel<<<(512 * 512 + 255) / 256, 256, 0, stream>>>(
        ow + (size_t)d * 512 * 512, wbuf, 512, 512, 1);
    gemm_wmma<<<dim3(SW_M / 128, 512 / 64), 256, 0, stream>>>(
        attn_o, wbuf, ob + d * 512, xw, xa, SW_M, 512, 512, /*epi*/1, 0);

    // LN2 -> h2 (f16)
    ln512_kernel<<<SW_M, 128, 0, stream>>>(xa, g2 + d * 512, b2 + d * 512,
                                           hbuf, nullptr, -1);

    // MLP1: gelu(h2 @ w1 + b1) -> gbuf (f16)
    cvt_kernel<<<(512 * 2048 + 255) / 256, 256, 0, stream>>>(
        w1 + (size_t)d * 512 * 2048, wbuf, 512, 2048, 0);
    gemm_wmma<<<dim3(SW_M / 128, 2048 / 64), 256, 0, stream>>>(
        hbuf, wbuf, bb1 + d * 2048, nullptr, gbuf, SW_M, 2048, 512, /*epi*/2, 0);

    // MLP2: g @ w2 + b2 + xa, un-window/un-shift permuted store -> ximg (f32)
    cvt_kernel<<<(2048 * 512 + 255) / 256, 256, 0, stream>>>(
        w2 + (size_t)d * 2048 * 512, wbuf, 2048, 512, 0);
    gemm_wmma<<<dim3(SW_M / 128, 512 / 64), 256, 0, stream>>>(
        gbuf, wbuf, bb2 + d * 512, xa, ximg, SW_M, 512, 2048, /*epi*/3, shift);
  }

  // PatchMerge: gather 2x2 -> LN(2048) -> @ merge_w^T -> d_out (f32)
  ln_merge_kernel<<<8192, 256, 0, stream>>>(ximg, mg, mb, hbuf);
  cvt_kernel<<<(2048 * 1024 + 255) / 256, 256, 0, stream>>>(
      mw, wbuf, 2048, 1024, 1);
  gemm_wmma<<<dim3(8192 / 128, 1024 / 64), 256, 0, stream>>>(
      hbuf, wbuf, nullptr, nullptr, d_out, 8192, 1024, 2048, /*epi*/4, 0);
}
